// PointConvStridePE_9165460209726
// MI455X (gfx1250) — matrix operand order repaired and checked by reading, not compiled, not run on hardware
//
#include <hip/hip_runtime.h>
#include <hip/hip_bf16.h>

typedef __attribute__((ext_vector_type(16))) _Float16 v16h;
typedef __attribute__((ext_vector_type(8)))  _Float16 v8h;
typedef __attribute__((ext_vector_type(8)))  float    v8f;

#define K_INV 0.99999500003749968f   // 1/sqrt(1+1e-5)

struct Params {
    const float *dense_xyz, *dense_feats, *dense_norm;
    const float *sparse_xyz, *sparse_norm;
    const int   *nei;
    const float *pe0_w,*pe0_b,*pe0_g,*pe0_bb;
    const float *pe1_w,*pe1_b,*pe1_g,*pe1_bb;
    const float *wn0_w,*wn0_b,*wn0_g,*wn0_bb;
    const float *wn1_w,*wn1_b,*wn1_g,*wn1_bb;
    const float *wn2_w,*wn2_b,*wn2_g,*wn2_bb;
    const float *lin_b,*lin_g,*lin_bb;
    const float *u2_w,*u2_b,*u2_g,*u2_bb;
    const float *sh_w,*sh_b,*sh_g,*sh_bb;
    const float    *feats_x;        // ws: [B*N][32] f32
    const _Float16 *lin_w_h;        // ws: [64][1024] f16 (pre-converted lin weight)
    float *out;                     // [B*M][128]
    float *vi_out;                  // [B*M][16][12]
    int N, M;
};

__device__ __forceinline__ float lrelu(float x) { return x > 0.f ? x : 0.1f * x; }

// ---------------- Kernel 1: feats_x = leaky_relu(lbn(dense_feats, unary1)) ----------------
__global__ void __launch_bounds__(256)
feats_x_kernel(const float* __restrict__ feats,
               const float* __restrict__ w, const float* __restrict__ b,
               const float* __restrict__ g, const float* __restrict__ bb,
               float* __restrict__ out, int rows) {
    int gid = blockIdx.x * 256 + threadIdx.x;
    int row = gid >> 5, oc = gid & 31;
    if (row >= rows) return;
    const float* f  = feats + (size_t)row * 64;
    const float* wr = w + oc * 64;
    float acc = b[oc];
    #pragma unroll
    for (int c = 0; c < 64; ++c) acc += f[c] * wr[c];
    float y = acc * (g[oc] * K_INV) + bb[oc];
    out[(size_t)row * 32 + oc] = lrelu(y);
}

// ---------------- Kernel 1b: one-time f32 -> f16 conversion of lin weight ----------------
__global__ void __launch_bounds__(256)
cvt_f16_kernel(const float* __restrict__ in, _Float16* __restrict__ out, int n) {
    int i = blockIdx.x * 256 + threadIdx.x;
    if (i < n) out[i] = (_Float16)in[i];
}

// ---------------- Kernel 2: fused tile kernel (32 sparse points / workgroup) ----------------
// LDS layout (dynamic):
//   [0,       65536)  Eh   : _Float16 [32][1024]   einsum result (A matrix of lin GEMM)
//   [65536,  131072)  cat  : _Float16 [32][16][64] (aliased later by lin_out/maxf)
//       alias: lin_out float[32][64] @65536 ; maxf float[32][64] @73728
//   [131072, 147456)  wt_h : _Float16 [32][16][16]
//   [147456, 149504)  nei_s: int      [32][16]
//   [149504, 149888)  sxyz : float    [32][3]
//   [149888, 150272)  snorm: float    [32][3]
#define SMEM_BYTES 150272

__global__ void __launch_bounds__(256)
pointconv_tile_kernel(Params P) {
    extern __shared__ char smem[];
    _Float16* Eh      = (_Float16*)(smem);
    _Float16* cat     = (_Float16*)(smem + 65536);
    float*    lin_out = (float*)   (smem + 65536);
    float*    maxf    = (float*)   (smem + 65536 + 8192);
    _Float16* wt_h    = (_Float16*)(smem + 131072);
    int*      nei_s   = (int*)     (smem + 147456);
    float*    sxyz    = (float*)   (smem + 149504);
    float*    snorm   = (float*)   (smem + 149888);

    const int tid = threadIdx.x;
    const int tile_base = blockIdx.x * 32;          // flattened b*M+m
    const int N = P.N, M = P.M;

    // ---- stage indices and sparse coords ----
    #pragma unroll
    for (int it = 0; it < 2; ++it) {
        int i = tid + it * 256;                     // 0..511 = (lm,k)
        nei_s[i] = P.nei[(size_t)tile_base * 16 + i];
    }
    if (tid < 96) {                                 // 32 points * 3 comps
        int lm = tid / 3, c = tid % 3;
        size_t bm = tile_base + lm;
        sxyz[lm * 3 + c]  = P.sparse_xyz[bm * 3 + c];
        snorm[lm * 3 + c] = P.sparse_norm[bm * 3 + c];
    }
    __syncthreads();

    // ---- phase 1: per (m,k): vi, PE chain, weightnet chain, cat ----
    for (int it = 0; it < 2; ++it) {
        int i = tid + it * 256;
        int lm = i >> 4, k = i & 15;
        int bm = tile_base + lm;
        int b  = bm / M;
        int idx = nei_s[lm * 16 + k];
        const float* gx = P.dense_xyz  + ((size_t)b * N + idx) * 3;
        const float* gn = P.dense_norm + ((size_t)b * N + idx) * 3;
        float sx = sxyz[lm*3+0], sy = sxyz[lm*3+1], sz = sxyz[lm*3+2];
        float nx = snorm[lm*3+0], ny = snorm[lm*3+1], nz = snorm[lm*3+2];
        float rx = gx[0] - sx, ry = gx[1] - sy, rz = gx[2] - sz;
        float gnx = gn[0], gny = gn[1], gnz = gn[2];

        float rn = sqrtf(rx*rx + ry*ry + rz*rz);
        float ir = 1.f / fmaxf(rn, 1e-12f);
        float hx = rx*ir, hy = ry*ir, hz = rz*ir;            // r_hat
        float proj = nx*hx + ny*hy + nz*hz;
        float vx = nx - proj*hx, vy = ny - proj*hy, vz = nz - proj*hz;
        float vl = sqrtf(vx*vx + vy*vy + vz*vz);
        float iv = 1.f / fmaxf(vl, 1e-12f);
        vx *= iv; vy *= iv; vz *= iv;                        // v_miu
        float wx = hy*vz - hz*vy, wy = hz*vx - hx*vz, wz = hx*vy - hy*vx;
        float wl = sqrtf(wx*wx + wy*wy + wz*wz);
        float iw = 1.f / fmaxf(wl, 1e-12f);
        wx *= iw; wy *= iw; wz *= iw;                        // w_miu
        float t1 = gnx*nx + gny*ny + gnz*nz;
        float t3 = hx*gnx + hy*gny + hz*gnz;
        float t4 = rx*nx + ry*ny + rz*nz;
        float t6 = gnx*vx + gny*vy + gnz*vz;
        float t7 = gnx*wx + gny*wy + gnz*wz;
        float cx = gny*nz - gnz*ny, cy = gnz*nx - gnx*nz, cz = gnx*ny - gny*nx;
        float t8 = rx*cx + ry*cy + rz*cz;

        float vi[12] = {t1, proj, t3, t4, t3, t6, t7, t8, rn, rx, ry, rz};
        float* vo = P.vi_out + ((size_t)bm * 16 + k) * 12;
        #pragma unroll
        for (int j = 0; j < 12; ++j) vo[j] = vi[j];

        // PE: 3 -> 32 -> 32 (relu)
        float h0[32];
        #pragma unroll
        for (int o = 0; o < 32; ++o) {
            float a = P.pe0_b[o] + P.pe0_w[o*3+0]*rx + P.pe0_w[o*3+1]*ry + P.pe0_w[o*3+2]*rz;
            a = a * (P.pe0_g[o] * K_INV) + P.pe0_bb[o];
            h0[o] = fmaxf(a, 0.f);
        }
        float h1[32];
        for (int o = 0; o < 32; ++o) {
            const float* wr = P.pe1_w + o * 32;
            float a = P.pe1_b[o];
            #pragma unroll
            for (int c = 0; c < 32; ++c) a += wr[c] * h0[c];
            a = a * (P.pe1_g[o] * K_INV) + P.pe1_bb[o];
            h1[o] = fmaxf(a, 0.f);
        }

        // weightnet: 12 -> 8 -> 8 -> 16 (relu)
        float a0[8];
        #pragma unroll
        for (int o = 0; o < 8; ++o) {
            const float* wr = P.wn0_w + o * 12;
            float a = P.wn0_b[o];
            #pragma unroll
            for (int c = 0; c < 12; ++c) a += wr[c] * vi[c];
            a = a * (P.wn0_g[o] * K_INV) + P.wn0_bb[o];
            a0[o] = fmaxf(a, 0.f);
        }
        float a1[8];
        #pragma unroll
        for (int o = 0; o < 8; ++o) {
            const float* wr = P.wn1_w + o * 8;
            float a = P.wn1_b[o];
            #pragma unroll
            for (int c = 0; c < 8; ++c) a += wr[c] * a0[c];
            a = a * (P.wn1_g[o] * K_INV) + P.wn1_bb[o];
            a1[o] = fmaxf(a, 0.f);
        }
        _Float16* wtr = wt_h + (lm * 16 + k) * 16;
        #pragma unroll
        for (int o = 0; o < 16; ++o) {
            const float* wr = P.wn2_w + o * 8;
            float a = P.wn2_b[o];
            #pragma unroll
            for (int c = 0; c < 8; ++c) a += wr[c] * a1[c];
            a = a * (P.wn2_g[o] * K_INV) + P.wn2_bb[o];
            wtr[o] = (_Float16)fmaxf(a, 0.f);
        }

        // cat = [gathered feats_x (32) | feat_pe (32)] as f16
        const float* fx = P.feats_x + ((size_t)b * N + idx) * 32;
        _Float16* cr = cat + (lm * 16 + k) * 64;
        #pragma unroll
        for (int c = 0; c < 32; ++c) cr[c] = (_Float16)fx[c];
        #pragma unroll
        for (int c = 0; c < 32; ++c) cr[32 + c] = (_Float16)h1[c];
    }
    __syncthreads();

    // ---- phase 2: einsum  E[m][c*16+o] = sum_k cat[m][k][c] * wt[m][k][o] ----
    for (int it = 0; it < 8; ++it) {
        int p = tid + it * 256;                     // 0..2047 = (m,c)
        int m = p >> 6, c = p & 63;
        float acc[16];
        #pragma unroll
        for (int o = 0; o < 16; ++o) acc[o] = 0.f;
        #pragma unroll
        for (int k = 0; k < 16; ++k) {
            float a = (float)cat[(m * 16 + k) * 64 + c];
            const _Float16* wr = wt_h + (m * 16 + k) * 16;
            #pragma unroll
            for (int o = 0; o < 16; ++o) acc[o] += a * (float)wr[o];
        }
        _Float16* er = Eh + m * 1024 + c * 16;
        #pragma unroll
        for (int o = 0; o < 16; ++o) er[o] = (_Float16)acc[o];
    }
    __syncthreads();

    // ---- phase 3: lin GEMM (32x1024)@(1024x64) via v_wmma_f32_16x16x32_f16 ----
    {
        int wv = tid >> 5, lane = tid & 31;
        int g = wv >> 2, nt = wv & 3;               // row-group, n-tile
        int row = lane & 15;
        bool hi = lane >= 16;
        int n_ch = nt * 16 + row;                   // B: column = lane%16
        const _Float16* wrow = P.lin_w_h + (size_t)n_ch * 1024;
        const _Float16* arow = Eh + (g * 16 + row) * 1024;
        v8f acc = {};
        #pragma unroll 4
        for (int kt = 0; kt < 32; ++kt) {
            // A frag: lanes 0-15 hold K in {kb..kb+7} U {kb+16..kb+23}, kb = kt*32 (+8 hi half)
            int kb = kt * 32 + (hi ? 8 : 0);
            v8h alo = *(const v8h*)(arow + kb);
            v8h ahi = *(const v8h*)(arow + kb + 16);
            v16h A = __builtin_shufflevector(alo, ahi, 0,1,2,3,4,5,6,7,8,9,10,11,12,13,14,15);
            // B frag: per lane K in {kb..kb+7} U {kb+16..kb+23} of its column (pre-converted), same as A
            int kbb = kt * 32 + (hi ? 8 : 0);
            v8h blo = *(const v8h*)(wrow + kbb);
            v8h bhi = *(const v8h*)(wrow + kbb + 16);
            v16h Bv = __builtin_shufflevector(blo, bhi, 0,1,2,3,4,5,6,7,8,9,10,11,12,13,14,15);
            acc = __builtin_amdgcn_wmma_f32_16x16x32_f16(false, A, false, Bv,
                                                         (short)0, acc, false, false);
        }
        // writeback D (16x16, f32): VGPR r -> M=r (lanes 0-15) / M=8+r (lanes 16-31), N=lane%16
        float sc = P.lin_g[n_ch] * K_INV;
        float bi = P.lin_b[n_ch];
        float bb = P.lin_bb[n_ch];
        #pragma unroll
        for (int r = 0; r < 8; ++r) {
            int mloc = g * 16 + r + (hi ? 8 : 0);
            lin_out[mloc * 64 + n_ch] = lrelu((acc[r] + bi) * sc + bb);
        }
    }

    // ---- phase 3.5: max-pool of raw dense_feats over K (aliases cat region) ----
    for (int it = 0; it < 8; ++it) {
        int p = tid + it * 256;                     // (m,c)
        int m = p >> 6, c = p & 63;
        int bm = tile_base + m;
        int b = bm / M;
        float mx = -3.402823466e38f;
        #pragma unroll
        for (int k = 0; k < 16; ++k) {
            int idx = nei_s[m * 16 + k];
            mx = fmaxf(mx, P.dense_feats[((size_t)b * N + idx) * 64 + c]);
        }
        maxf[m * 64 + c] = mx;
    }
    __syncthreads();

    // ---- phase 4: unary2(64->128) + shortcut(64->128) + add + leaky ----
    for (int it = 0; it < 16; ++it) {
        int p = tid + it * 256;                     // (m,j)
        int m = p >> 7, j = p & 127;
        const float* u2r = P.u2_w + j * 64;
        const float* shr = P.sh_w + j * 64;
        float a = P.u2_b[j], s = P.sh_b[j];
        #pragma unroll
        for (int c = 0; c < 64; ++c) {
            a += u2r[c] * lin_out[m * 64 + c];
            s += shr[c] * maxf[m * 64 + c];
        }
        a = a * (P.u2_g[j] * K_INV) + P.u2_bb[j];
        s = s * (P.sh_g[j] * K_INV) + P.sh_bb[j];
        int bm = tile_base + m;
        P.out[(size_t)bm * 128 + j] = lrelu(a + s);
    }
}

extern "C" void kernel_launch(void* const* d_in, const int* in_sizes, int n_in,
                              void* d_out, int out_size, void* d_ws, size_t ws_size,
                              hipStream_t stream) {
    const int B = 2;
    const int N = in_sizes[1] / (64 * B);           // dense_feats (B,N,64)
    const int M = in_sizes[5] / (16 * B);           // nei_inds (B,M,16)

    const float* dense_feats = (const float*)d_in[1];

    // workspace carve: feats_x [B*N][32] f32, then lin_w f16 [64*1024]
    float*     feats_x = (float*)d_ws;
    size_t     fx_bytes = (size_t)B * N * 32 * sizeof(float);
    _Float16*  lin_w_h = (_Float16*)((char*)d_ws + fx_bytes);

    // unary1 precompute over all dense points
    {
        int rows = B * N;
        int threads = rows * 32;
        feats_x_kernel<<<(threads + 255) / 256, 256, 0, stream>>>(
            dense_feats,
            (const float*)d_in[6], (const float*)d_in[7],
            (const float*)d_in[8], (const float*)d_in[9],
            feats_x, rows);
    }
    // lin weight f32 -> f16 (once)
    {
        int n = in_sizes[30];                       // 64*1024
        cvt_f16_kernel<<<(n + 255) / 256, 256, 0, stream>>>(
            (const float*)d_in[30], lin_w_h, n);
    }

    Params P;
    P.dense_xyz   = (const float*)d_in[0];
    P.dense_feats = dense_feats;
    P.dense_norm  = (const float*)d_in[2];
    P.sparse_xyz  = (const float*)d_in[3];
    P.sparse_norm = (const float*)d_in[4];
    P.nei         = (const int*)  d_in[5];
    P.pe0_w=(const float*)d_in[10]; P.pe0_b=(const float*)d_in[11]; P.pe0_g=(const float*)d_in[12]; P.pe0_bb=(const float*)d_in[13];
    P.pe1_w=(const float*)d_in[14]; P.pe1_b=(const float*)d_in[15]; P.pe1_g=(const float*)d_in[16]; P.pe1_bb=(const float*)d_in[17];
    P.wn0_w=(const float*)d_in[18]; P.wn0_b=(const float*)d_in[19]; P.wn0_g=(const float*)d_in[20]; P.wn0_bb=(const float*)d_in[21];
    P.wn1_w=(const float*)d_in[22]; P.wn1_b=(const float*)d_in[23]; P.wn1_g=(const float*)d_in[24]; P.wn1_bb=(const float*)d_in[25];
    P.wn2_w=(const float*)d_in[26]; P.wn2_b=(const float*)d_in[27]; P.wn2_g=(const float*)d_in[28]; P.wn2_bb=(const float*)d_in[29];
    P.lin_b=(const float*)d_in[31]; P.lin_g=(const float*)d_in[32]; P.lin_bb=(const float*)d_in[33];
    P.u2_w =(const float*)d_in[34]; P.u2_b =(const float*)d_in[35]; P.u2_g =(const float*)d_in[36]; P.u2_bb =(const float*)d_in[37];
    P.sh_w =(const float*)d_in[38]; P.sh_b =(const float*)d_in[39]; P.sh_g =(const float*)d_in[40]; P.sh_bb =(const float*)d_in[41];
    P.feats_x = feats_x;
    P.lin_w_h = lin_w_h;
    P.out     = (float*)d_out;
    P.vi_out  = (float*)d_out + (size_t)B * M * 128;
    P.N = N; P.M = M;

    int tiles = (B * M + 31) / 32;
    pointconv_tile_kernel<<<tiles, 256, SMEM_BYTES, stream>>>(P);
}
